// Dynamics_23828478558279
// MI455X (gfx1250) — matrix operand order repaired and checked by reading, not compiled, not run on hardware
//
#include <hip/hip_runtime.h>
#include <hip/hip_bf16.h>
#include <math.h>

// ---------------------------------------------------------------------------
// Problem constants (from reference): B=2, T=4096, D=128, K=256
// ---------------------------------------------------------------------------
#define BB 2
#define TT 4096
#define DD 128
#define KK 256
#define NROW (BB * TT)        // 8192 rows
#define NELEM (NROW * DD)     // 1048576 elements

typedef _Float16 v16h __attribute__((ext_vector_type(16)));
typedef float    v8f  __attribute__((ext_vector_type(8)));

union V16U { v16h v; uint4 q[2]; };

static __device__ __forceinline__ v8f wmma_f16(v16h a, v16h b, v8f c) {
    return __builtin_amdgcn_wmma_f32_16x16x32_f16(
        /*neg_a=*/false, a, /*neg_b=*/false, b,
        /*c_mod=*/(short)0, c, /*reuse_a=*/false, /*reuse_b=*/false);
}

// ---------------------------------------------------------------------------
// Kernel 1: depthwise causal conv  memory_force[b,t,d] = sum_k z[b,t-255+k,d]*w[k,d] + cb[d]
// ---------------------------------------------------------------------------
__global__ void conv_kernel(const float* __restrict__ z,
                            const float* __restrict__ ck,   // [K][1][D]
                            const float* __restrict__ cb,   // [D]
                            float* __restrict__ mf) {
    int tid = blockIdx.x * blockDim.x + threadIdx.x;
    if (tid >= NELEM) return;
    int d = tid & (DD - 1);
    int t = (tid >> 7) & (TT - 1);
    int b = tid >> 19;
    const float* zb = z + (size_t)b * TT * DD;
    float acc = cb[d];
    int k0 = (t >= KK - 1) ? 0 : (KK - 1 - t);
#pragma unroll 4
    for (int k = k0; k < KK; ++k) {
        acc += zb[(size_t)(t - (KK - 1) + k) * DD + d] * ck[k * DD + d];
    }
    mf[tid] = acc;
}

// ---------------------------------------------------------------------------
// Kernel 2: projection + sphere-normalize.  outH = f16(normalize(zin@W + bias))
// Optionally also writes transposed copy outT[b][d][t] (for k_hat).
// One block = 16 rows, 256 threads.
// ---------------------------------------------------------------------------
__global__ void __launch_bounds__(256)
proj_norm_kernel(const float* __restrict__ zin,
                 const float* __restrict__ W,      // [D][D] row-major (d_in, d_out)
                 const float* __restrict__ bias,   // [D]
                 _Float16* __restrict__ outH,      // [NROW][D]
                 _Float16* __restrict__ outT) {    // [B][D][T] or nullptr
    __shared__ float zs[16][DD];
    __shared__ float ps[16][DD + 1];
    __shared__ float invn[16];

    const int r0   = blockIdx.x * 16;
    const int col  = threadIdx.x & (DD - 1);
    const int half = threadIdx.x >> 7;   // 0..1

#pragma unroll
    for (int i = 0; i < 8; ++i) {
        int r = half * 8 + i;
        zs[r][col] = zin[(size_t)(r0 + r) * DD + col];
    }
    __syncthreads();

    float acc[8];
    float bv = bias[col];
#pragma unroll
    for (int i = 0; i < 8; ++i) acc[i] = bv;
#pragma unroll 8
    for (int k = 0; k < DD; ++k) {
        float w = W[k * DD + col];
#pragma unroll
        for (int i = 0; i < 8; ++i) acc[i] += zs[half * 8 + i][k] * w;
    }
#pragma unroll
    for (int i = 0; i < 8; ++i) ps[half * 8 + i][col] = acc[i];
    __syncthreads();

    if (threadIdx.x < 16) {
        float s = 0.f;
#pragma unroll 8
        for (int c = 0; c < DD; ++c) { float v = ps[threadIdx.x][c]; s += v * v; }
        invn[threadIdx.x] = 1.0f / fmaxf(sqrtf(s), 1e-8f);
    }
    __syncthreads();

#pragma unroll
    for (int i = 0; i < 8; ++i) {
        int r = half * 8 + i;
        float h = ps[r][col] * invn[r];
        int g = r0 + r;
        outH[(size_t)g * DD + col] = (_Float16)h;
        if (outT) {
            int b = g >> 12;           // / TT
            int t = g & (TT - 1);
            outT[((size_t)(b * DD + col)) * TT + t] = (_Float16)h;
        }
    }
}

// ---------------------------------------------------------------------------
// Kernel 3: WoT[j][i] = f16(Wo[i][j])
// ---------------------------------------------------------------------------
__global__ void wot_kernel(const float* __restrict__ Wo, _Float16* __restrict__ WoT) {
    int tid = blockIdx.x * blockDim.x + threadIdx.x;
    if (tid >= DD * DD) return;
    int i = tid >> 7, j = tid & (DD - 1);
    WoT[(size_t)j * DD + i] = (_Float16)Wo[(size_t)i * DD + j];
}

// ---------------------------------------------------------------------------
// Kernel 4: causal sin-attention + fused @Wo.
//   total[q][:] = (sin(q_hat@k_hat^T) .* tril) @ k_hat @ Wo + bo + memory_force
// One block = one 16-query tile, 4 waves split the key range (32 keys/block).
// ---------------------------------------------------------------------------
__global__ void __launch_bounds__(128)
attn_kernel(const _Float16* __restrict__ qh,   // [NROW][D]
            const _Float16* __restrict__ kh,   // [NROW][D]
            const _Float16* __restrict__ khT,  // [B][D][T]
            const _Float16* __restrict__ WoT,  // [D][D] (out,in)
            const float* __restrict__ bo,      // [D]
            const float* __restrict__ mf,      // [NROW][D]
            float* __restrict__ total) {       // [NROW][D]
    __shared__ __align__(16) _Float16 attnS[4][16][40];     // per-wave attn tile, padded
    __shared__ __align__(16) float    partialS[3][16][132]; // cross-wave acc reduce
    __shared__ __align__(16) _Float16 accHS[16][136];       // f16 A for @Wo GEMM

    const int tile = blockIdx.x;
    const int R0   = tile * 16;            // global row base
    const int b    = R0 >> 12;             // / TT
    const int t0   = R0 & (TT - 1);
    const int lane = threadIdx.x & 31;
    // wave id is uniform across the wave: pin it to an SGPR so the key loop is
    // a uniform (s_cbranch) loop -> EXEC provably all-1s at every WMMA site.
    const int w    = __builtin_amdgcn_readfirstlane((int)(threadIdx.x >> 5));
    const int mr   = lane & 15;
    const int hi   = lane >> 4;

    // --- A operand: q_hat tile, 16-bit A, contiguous K order (lane=M, k = hi*16+e, same as B)
    v16h Aq[4];
    {
        const _Float16* qp = qh + (size_t)(R0 + mr) * DD;
#pragma unroll
        for (int c = 0; c < 4; ++c) {
            V16U u;
            u.q[0] = *(const uint4*)(qp + c * 32 + hi * 16);
            u.q[1] = *(const uint4*)(qp + c * 32 + hi * 16 + 8);
            Aq[c] = u.v;
        }
    }

    v8f acc[8];
#pragma unroll
    for (int dt = 0; dt < 8; ++dt)
#pragma unroll
        for (int j = 0; j < 8; ++j) acc[dt][j] = 0.0f;

    const int nkb = (t0 + 16 + 31) >> 5;   // causal #32-key blocks
    for (int kb = w; kb < nkb; kb += 4) {
        const int sbase = kb * 32;
        // ---- GEMM1: S = q_hat @ k_hat^T (two 16-key subtiles), sin+mask -> LDS
#pragma unroll
        for (int kt = 0; kt < 2; ++kt) {
            v8f S;
#pragma unroll
            for (int j = 0; j < 8; ++j) S[j] = 0.0f;
            const int key_t = sbase + kt * 16 + mr;     // B: lane = N = key
            const _Float16* kp = kh + ((size_t)(b * TT + key_t)) * DD;
#pragma unroll
            for (int c = 0; c < 4; ++c) {
                V16U u;                                   // B: elems = K = hi*16+e
                u.q[0] = *(const uint4*)(kp + c * 32 + hi * 16);
                u.q[1] = *(const uint4*)(kp + c * 32 + hi * 16 + 8);
                S = wmma_f16(Aq[c], u.v, S);
            }
#pragma unroll
            for (int j = 0; j < 8; ++j) {
                int qrow = t0 + j + 8 * hi;               // C: M = j + 8*hi
                float m = (sbase + kt * 16 + mr <= qrow) ? 1.0f : 0.0f;
                // dots are q_hat.k_hat of unit vectors: |x| <= ~1, so the
                // native (no range reduction) sin path is exact enough.
                float a = __sinf(S[j]) * m;
                attnS[w][j + 8 * hi][kt * 16 + mr] = (_Float16)a;
            }
        }
        // ---- GEMM2: acc += attn @ k_hat (K = 32 keys), B from k_hat^T (contig)
        v16h Aa;
        {
            V16U u;
            u.q[0] = *(const uint4*)&attnS[w][mr][hi * 16];
            u.q[1] = *(const uint4*)&attnS[w][mr][hi * 16 + 8];
            Aa = u.v;
        }
#pragma unroll
        for (int dt = 0; dt < 8; ++dt) {
            const _Float16* tp = khT + ((size_t)(b * DD + dt * 16 + mr)) * TT
                                 + sbase + hi * 16;
            V16U u;
            u.q[0] = *(const uint4*)(tp);
            u.q[1] = *(const uint4*)(tp + 8);
            acc[dt] = wmma_f16(Aa, u.v, acc[dt]);
        }
    }

    // ---- reduce partial accumulators across the 4 key-splitting waves
    if (w > 0) {
#pragma unroll
        for (int dt = 0; dt < 8; ++dt)
#pragma unroll
            for (int j = 0; j < 8; ++j)
                partialS[w - 1][j + 8 * hi][dt * 16 + mr] = acc[dt][j];
    }
    __syncthreads();

    if (w == 0) {
#pragma unroll
        for (int dt = 0; dt < 8; ++dt)
#pragma unroll
            for (int j = 0; j < 8; ++j) {
                int rr = j + 8 * hi, cc = dt * 16 + mr;
                float s = acc[dt][j] + partialS[0][rr][cc]
                        + partialS[1][rr][cc] + partialS[2][rr][cc];
                accHS[rr][cc] = (_Float16)s;
            }
        // ---- GEMM3: total = accsum @ Wo + bo + memory_force
        v16h Ac[4];
#pragma unroll
        for (int c = 0; c < 4; ++c) {
            V16U u;
            u.q[0] = *(const uint4*)&accHS[mr][c * 32 + hi * 16];
            u.q[1] = *(const uint4*)&accHS[mr][c * 32 + hi * 16 + 8];
            Ac[c] = u.v;
        }
#pragma unroll
        for (int dt = 0; dt < 8; ++dt) {
            v8f C;
#pragma unroll
            for (int j = 0; j < 8; ++j) C[j] = 0.0f;
            const _Float16* wp = WoT + (size_t)(dt * 16 + mr) * DD;  // B: lane = N = d_out
#pragma unroll
            for (int c = 0; c < 4; ++c) {
                V16U u;
                u.q[0] = *(const uint4*)(wp + c * 32 + hi * 16);
                u.q[1] = *(const uint4*)(wp + c * 32 + hi * 16 + 8);
                C = wmma_f16(Ac[c], u.v, C);
            }
            float bov = bo[dt * 16 + mr];
#pragma unroll
            for (int j = 0; j < 8; ++j) {
                size_t gi = (size_t)(R0 + j + 8 * hi) * DD + dt * 16 + mr;
                total[gi] = C[j] + bov + mf[gi];
            }
        }
    }
}

// ---------------------------------------------------------------------------
// Kernel 5: RK4 stage: tangent projection + v accumulation + sphere retract.
// One block = one row (128 threads).
// ---------------------------------------------------------------------------
static __device__ __forceinline__ float block_reduce_128(float v, float* sbuf) {
    int t = threadIdx.x;
    sbuf[t] = v;
    __syncthreads();
#pragma unroll
    for (int s = 64; s > 0; s >>= 1) {
        if (t < s) sbuf[t] += sbuf[t + s];
        __syncthreads();
    }
    float r = sbuf[0];
    __syncthreads();
    return r;
}

__global__ void __launch_bounds__(128)
rk4_stage_kernel(const float* __restrict__ total,
                 const float* __restrict__ zbase,
                 const float* __restrict__ zcur,
                 float* __restrict__ vacc,
                 float* __restrict__ outz,
                 int stage) {
    __shared__ float sbuf[128];
    size_t idx = (size_t)blockIdx.x * DD + threadIdx.x;
    float tv = total[idx];
    float zc = zcur[idx];
    float zb = zbase[idx];
    float dot = block_reduce_128(tv * zc, sbuf);
    float kd = tv - dot * zc;   // tangent projection

    float zn;
    if (stage == 1) {
        vacc[idx] = kd;
        zn = zb + 0.5f * kd;
    } else if (stage == 2) {
        vacc[idx] += 2.0f * kd;
        zn = zb + 0.5f * kd;
    } else if (stage == 3) {
        vacc[idx] += 2.0f * kd;
        zn = zb + kd;
    } else {
        float v = (vacc[idx] + kd) * (1.0f / 6.0f);
        zn = zb + v;
    }
    float n2 = block_reduce_128(zn * zn, sbuf);
    float inv = 1.0f / fmaxf(sqrtf(n2), 1e-8f);
    outz[idx] = zn * inv;
}

// ---------------------------------------------------------------------------
// Host launcher
// ---------------------------------------------------------------------------
extern "C" void kernel_launch(void* const* d_in, const int* in_sizes, int n_in,
                              void* d_out, int out_size, void* d_ws, size_t ws_size,
                              hipStream_t stream) {
    (void)in_sizes; (void)n_in; (void)out_size; (void)ws_size;
    const float* z  = (const float*)d_in[0];
    const float* ck = (const float*)d_in[1];
    const float* cb = (const float*)d_in[2];
    const float* Wq = (const float*)d_in[3];
    const float* bq = (const float*)d_in[4];
    const float* Wk = (const float*)d_in[5];
    const float* bk = (const float*)d_in[6];
    const float* Wo = (const float*)d_in[7];
    const float* bo = (const float*)d_in[8];
    float* out = (float*)d_out;

    // workspace carve-up (256B aligned)
    char* base = (char*)d_ws;
    size_t off = 0;
    auto carve = [&](size_t bytes) -> char* {
        char* p = base + off;
        off = (off + bytes + 255) & ~(size_t)255;
        return p;
    };
    float*     mf    = (float*)    carve((size_t)NELEM * 4);
    float*     totb  = (float*)    carve((size_t)NELEM * 4);
    float*     vacc  = (float*)    carve((size_t)NELEM * 4);
    float*     zcA   = (float*)    carve((size_t)NELEM * 4);
    float*     zcB   = (float*)    carve((size_t)NELEM * 4);
    _Float16*  qh    = (_Float16*) carve((size_t)NELEM * 2);
    _Float16*  kh    = (_Float16*) carve((size_t)NELEM * 2);
    _Float16*  khT   = (_Float16*) carve((size_t)NELEM * 2);
    _Float16*  WoT   = (_Float16*) carve((size_t)DD * DD * 2);

    // one-time precompute
    conv_kernel<<<NELEM / 256, 256, 0, stream>>>(z, ck, cb, mf);
    proj_norm_kernel<<<NROW / 16, 256, 0, stream>>>(z, Wk, bk, kh, khT);
    wot_kernel<<<(DD * DD) / 256, 256, 0, stream>>>(Wo, WoT);

    const int AT = NROW / 16;   // 512 attention tiles

    // stage 1 (z_curr = z)
    proj_norm_kernel<<<NROW / 16, 256, 0, stream>>>(z, Wq, bq, qh, nullptr);
    attn_kernel<<<AT, 128, 0, stream>>>(qh, kh, khT, WoT, bo, mf, totb);
    rk4_stage_kernel<<<NROW, 128, 0, stream>>>(totb, z, z, vacc, zcA, 1);

    // stage 2
    proj_norm_kernel<<<NROW / 16, 256, 0, stream>>>(zcA, Wq, bq, qh, nullptr);
    attn_kernel<<<AT, 128, 0, stream>>>(qh, kh, khT, WoT, bo, mf, totb);
    rk4_stage_kernel<<<NROW, 128, 0, stream>>>(totb, z, zcA, vacc, zcB, 2);

    // stage 3
    proj_norm_kernel<<<NROW / 16, 256, 0, stream>>>(zcB, Wq, bq, qh, nullptr);
    attn_kernel<<<AT, 128, 0, stream>>>(qh, kh, khT, WoT, bo, mf, totb);
    rk4_stage_kernel<<<NROW, 128, 0, stream>>>(totb, z, zcB, vacc, zcA, 3);

    // stage 4 -> final output
    proj_norm_kernel<<<NROW / 16, 256, 0, stream>>>(zcA, Wq, bq, qh, nullptr);
    attn_kernel<<<AT, 128, 0, stream>>>(qh, kh, khT, WoT, bo, mf, totb);
    rk4_stage_kernel<<<NROW, 128, 0, stream>>>(totb, z, zcA, vacc, out, 4);
}
